// RWKV_Tmix_x070_78769700209138
// MI455X (gfx1250) — compile-verified, matrix operand order checked by reading, not run on hardware
//
#include <hip/hip_runtime.h>
#include <hip/hip_bf16.h>

#define BB 4
#define TT 1024
#define CC 1024
#define HH 16
#define NN 64
#define MM (BB*TT)                 // 4096 rows
#define EPSV (1e-5f*64.0f)
#define LDSTRIDE 40                // 32 + 8 bf16 pad (80B rows -> conflict-free b128 reads)

typedef __bf16 bf16;
typedef __attribute__((ext_vector_type(16))) __bf16 v16bf;
typedef __attribute__((ext_vector_type(8)))  __bf16 v8bf;
typedef __attribute__((ext_vector_type(8)))  float  v8f;

__device__ __forceinline__ float sigm(float x){ return 1.f/(1.f+__expf(-x)); }

// Async 16-byte global -> LDS copy (CDNA5 GLOBAL_LOAD_ASYNC_TO_LDS_B128,
// tracked by ASYNCcnt). LDS dest passed as byte offset (low 32 bits of the
// generic shared-space address, per ISA LDS-aperture truncation).
__device__ __forceinline__ void async_copy16(bf16* lds_dst, const bf16* gsrc){
    unsigned lofs = (unsigned)(unsigned long long)(uintptr_t)lds_dst;
    unsigned long long ga = (unsigned long long)(uintptr_t)gsrc;
    asm volatile("global_load_async_to_lds_b128 %0, %1, off"
                 :: "v"(lofs), "v"(ga) : "memory");
}
__device__ __forceinline__ void async_wait0(){
    asm volatile("s_wait_asynccnt 0x0" ::: "memory");
}

// ---------------------------------------------------------------- fp32 -> bf16
__global__ void cvt_f32_bf16(const float* __restrict__ src, bf16* __restrict__ dst, int n){
    int i = blockIdx.x*blockDim.x + threadIdx.x;
    if (i < n) dst[i] = (bf16)src[i];
}

// Build one 16x32 bf16 A fragment (ISA 16-bit A layout) with the time-shift mix
// fused in: mix = x + (tm*prev - x)*mu, tm = (t>0). Branch-free: prev address is
// clamped to row m when t==0, tm zeroes its contribution.
__device__ __forceinline__ v16bf a_frag_mix(const float* __restrict__ x,
                                            const float* __restrict__ mua, // [16] mu slice
                                            int m, int k0, int kb)
{
    const int   t  = m & (TT-1);
    const float tm = (t > 0) ? 1.f : 0.f;
    const float* xp = x + (size_t)m*CC + k0 + kb;
    const float* pp = (t > 0) ? (xp - CC) : xp;   // always-valid address
    float xa[16], pa[16];
    *(float4*)(xa+ 0) = *(const float4*)(xp+ 0);
    *(float4*)(xa+ 4) = *(const float4*)(xp+ 4);
    *(float4*)(xa+ 8) = *(const float4*)(xp+16);
    *(float4*)(xa+12) = *(const float4*)(xp+20);
    *(float4*)(pa+ 0) = *(const float4*)(pp+ 0);
    *(float4*)(pa+ 4) = *(const float4*)(pp+ 4);
    *(float4*)(pa+ 8) = *(const float4*)(pp+16);
    *(float4*)(pa+12) = *(const float4*)(pp+20);
    v16bf a;
    #pragma unroll
    for (int e=0; e<16; ++e)
        a[e] = (bf16)(xa[e] + (tm*pa[e] - xa[e])*mua[e]);
    return a;
}

__device__ __forceinline__ v16bf a_frag_bf(const bf16* __restrict__ Abf,
                                           int m, int k0, int kb)
{
    const bf16* ap = Abf + (size_t)m*CC + k0 + kb;
    const v8bf lo = *(const v8bf*)(ap);
    const v8bf hi = *(const v8bf*)(ap + 16);
    v16bf a;
    #pragma unroll
    for (int e=0; e<8; ++e){ a[e] = lo[e]; a[e+8] = hi[e]; }
    return a;
}

// ---------------------------------------------------------------- WMMA GEMM
// out[M,C] = A[M,C] @ W^T, W stored [C_out, C_in] row-major (bf16).
// MODE 0: A built on the fly as x + (shift(x)-x)*mu   (fused time-shift mix)
// MODE 1: A read from a bf16 buffer.
// Block: 256 threads = 8 waves. Wave -> 32 rows x 64 cols (8 WMMA / K-step).
// Weight tile (64 cols x 32 K, shared by all 8 waves) double-buffered in LDS,
// staged with GLOBAL_LOAD_ASYNC_TO_LDS_B128 (ASYNCcnt).
template<int MODE>
__global__ void wmma_gemm(const float* __restrict__ x,   // [M,C] (MODE 0)
                          const float* __restrict__ mu,  // [C]   (MODE 0)
                          const bf16*  __restrict__ Abf, // [M,C] (MODE 1)
                          const bf16*  __restrict__ Wbf, // [C,C]
                          float* __restrict__ out)       // [M,C]
{
    __shared__ bf16 lB[2][64*LDSTRIDE];

    const int lane  = threadIdx.x & 31;
    const int wave  = threadIdx.x >> 5;
    const int tileM = blockIdx.y*256 + wave*32;
    const int tileN = blockIdx.x*64;
    const int l16   = lane & 15;
    const int kb    = (lane >> 4) * 8;     // K-half select per ISA 16-bit layout
    const int m0    = tileM + l16;
    const int m1    = m0 + 16;

    // cooperative weight staging: thread -> (row sn of 64, 8-elem chunk sc of 32)
    const int sn = threadIdx.x >> 2;
    const int sc = (threadIdx.x & 3) * 8;
    const bf16* wsrc = Wbf + (size_t)(tileN + sn)*CC + sc;
    bf16* ldst0 = &lB[0][sn*LDSTRIDE + sc];
    bf16* ldst1 = &lB[1][sn*LDSTRIDE + sc];

    v8f acc[2][4];
    #pragma unroll
    for (int rr=0; rr<2; ++rr)
        #pragma unroll
        for (int nn=0; nn<4; ++nn)
            #pragma unroll
            for (int e=0; e<8; ++e) acc[rr][nn][e] = 0.f;

    // prologue: stage K-slice 0 into buffer 0
    async_copy16(ldst0, wsrc);
    async_wait0();
    __syncthreads();

    for (int it=0; it<CC/32; ++it){
        const int k0  = it*32;
        const int cur = it & 1;
        if (it+1 < CC/32)   // prefetch next K-slice into the other buffer
            async_copy16(cur ? ldst0 : ldst1, wsrc + k0 + 32);

        v16bf a0, a1;
        if (MODE == 0){
            float mua[16];
            *(float4*)(mua+ 0) = *(const float4*)(mu+k0+kb+ 0);
            *(float4*)(mua+ 4) = *(const float4*)(mu+k0+kb+ 4);
            *(float4*)(mua+ 8) = *(const float4*)(mu+k0+kb+16);
            *(float4*)(mua+12) = *(const float4*)(mu+k0+kb+20);
            a0 = a_frag_mix(x, mua, m0, k0, kb);
            a1 = a_frag_mix(x, mua, m1, k0, kb);
        } else {
            a0 = a_frag_bf(Abf, m0, k0, kb);
            a1 = a_frag_bf(Abf, m1, k0, kb);
        }

        #pragma unroll
        for (int nn=0; nn<4; ++nn){
            const bf16* lp = &lB[cur][(nn*16 + l16)*LDSTRIDE + kb];
            const v8bf blo = *(const v8bf*)(lp);
            const v8bf bhi = *(const v8bf*)(lp + 16);
            v16bf bfrag;
            #pragma unroll
            for (int e=0; e<8; ++e){ bfrag[e] = blo[e]; bfrag[e+8] = bhi[e]; }
            acc[0][nn] = __builtin_amdgcn_wmma_f32_16x16x32_bf16(
                false, a0, false, bfrag, (short)0, acc[0][nn], false, false);
            acc[1][nn] = __builtin_amdgcn_wmma_f32_16x16x32_bf16(
                false, a1, false, bfrag, (short)0, acc[1][nn], false, false);
        }
        async_wait0();      // staged writes for next slice are in LDS
        __syncthreads();
    }

    // C/D layout: VGPR g -> row +g (lanes 0-15) / row +8+g (lanes 16-31)
    #pragma unroll
    for (int rr=0; rr<2; ++rr)
        #pragma unroll
        for (int nn=0; nn<4; ++nn)
            #pragma unroll
            for (int gi=0; gi<8; ++gi){
                const int row = tileM + rr*16 + gi + (lane>>4)*8;
                const int col = tileN + nn*16 + l16;
                out[(size_t)row*CC + col] = acc[rr][nn][gi];
            }
}

// ---------------------------------------------------------------- LoRA stage 1
// hid[m,d] = act( sum_k mix(x)[m,k] * W1[k,d] ), blockDim = D
template<int ACT>  // 0 none, 1 tanh, 2 sigmoid
__global__ void lora_stage1(const float* __restrict__ x, const float* __restrict__ mu,
                            const float* __restrict__ W1, int D,
                            float* __restrict__ hid)
{
    const int m = blockIdx.x;
    const int d = threadIdx.x;
    const int t = m & (TT-1);
    const float tm = (t > 0) ? 1.f : 0.f;
    const float* xr  = x + (size_t)m*CC;
    const float* xpr = (t > 0) ? (xr - CC) : xr;
    float acc = 0.f;
    for (int k=0; k<CC; ++k){
        float xv = xr[k];
        float pv = tm * xpr[k];
        acc += (xv + (pv - xv)*mu[k]) * W1[(size_t)k*D + d];
    }
    if (ACT==1) acc = tanhf(acc);
    else if (ACT==2) acc = sigm(acc);
    hid[(size_t)m*D + d] = acc;
}

// ---------------------------------------------------------------- LoRA stage 2
template<int POST> // 0 none, 1 sigmoid, 2 decay: exp(-exp(-softplus(-z)-0.5))
__global__ void lora_stage2(const float* __restrict__ hid, int D,
                            const float* __restrict__ W2,
                            const float* __restrict__ bias,
                            float* __restrict__ out)
{
    const int m = blockIdx.y;
    const int c = blockIdx.x*blockDim.x + threadIdx.x;
    const float* hp = hid + (size_t)m*D;
    float acc = bias ? bias[c] : 0.f;
    for (int d=0; d<D; ++d) acc += hp[d] * W2[(size_t)d*CC + c];
    if (POST==1) acc = sigm(acc);
    else if (POST==2){
        float nz = -acc;
        float sp = (nz > 20.f) ? nz : log1pf(__expf(nz));   // softplus(-z)
        float w  = -sp - 0.5f;
        acc = __expf(-__expf(w));                            // decay in (0,1)
    }
    out[(size_t)m*CC + c] = acc;
}

// ---------------------------------------------------------------- k/kk/v prep
// kk = normalize_head(k*k_k); k *= 1+(a-1)*k_a; v = v + (vfirst-v)*vgate
__global__ void kv_prep(float* __restrict__ k, float* __restrict__ v,
                        const float* __restrict__ a, const float* __restrict__ vgate,
                        const float* __restrict__ vfirst,
                        const float* __restrict__ k_k, const float* __restrict__ k_a,
                        float* __restrict__ kk)
{
    __shared__ float red[NN];
    const int mh = blockIdx.x;        // m*H + h
    const int m  = mh >> 4, h = mh & 15;
    const int tid = threadIdx.x;
    const int c   = h*NN + tid;
    const size_t g = (size_t)m*CC + c;
    float kv = k[g]*k_k[c];
    red[tid] = kv*kv;
    __syncthreads();
    for (int s=32; s>0; s>>=1){ if (tid<s) red[tid]+=red[tid+s]; __syncthreads(); }
    float denom = fmaxf(sqrtf(red[0]), 1e-12f);
    kk[g] = kv/denom;
    float av = a[g];
    k[g]  = k[g]*(1.f + (av-1.f)*k_a[c]);
    float vv = v[g];
    v[g]  = vv + (vfirst[g]-vv)*vgate[g];
}

// ---------------------------------------------------------------- sequential scan
// One block = ONE wave32 per (b,h); thread i owns state rows i and i+32
// (128 fp32 in registers). Single-wave workgroup -> barriers degenerate to
// waitcnts (no barrier resource allocated), which matters at 1024 steps.
__global__ void rwkv_scan(const float* __restrict__ r, const float* __restrict__ wd,
                          const float* __restrict__ k, const float* __restrict__ kk,
                          const float* __restrict__ a, const float* __restrict__ v,
                          const float* __restrict__ state0, float* __restrict__ y)
{
    __shared__ float lr[NN], lw[NN], lk[NN], lkk[NN], la[NN], lv[NN];
    const int bh = blockIdx.x;        // b*H + h
    const int b  = bh >> 4, h = bh & 15;
    const int i  = threadIdx.x;       // 0..31
    float s0[NN], s1[NN];
    const float* sp0 = state0 + ((size_t)bh*NN + i)*NN;
    const float* sp1 = sp0 + 32*NN;
    #pragma unroll
    for (int j=0; j<NN; ++j){ s0[j] = sp0[j]; s1[j] = sp1[j]; }

    for (int t=0; t<TT; ++t){
        const size_t g = ((size_t)(b*TT + t))*CC + h*NN;
        lr[i]=r[g+i];    lr[i+32]=r[g+i+32];
        lw[i]=wd[g+i];   lw[i+32]=wd[g+i+32];
        lk[i]=k[g+i];    lk[i+32]=k[g+i+32];
        lkk[i]=kk[g+i];  lkk[i+32]=kk[g+i+32];
        la[i]=a[g+i];    la[i+32]=a[g+i+32];
        lv[i]=v[g+i];    lv[i+32]=v[g+i+32];
        __syncthreads();
        const float vi0 = lv[i], vi1 = lv[i+32];
        float sa0 = 0.f, sa1 = 0.f;
        #pragma unroll
        for (int j=0; j<NN; ++j){
            const float kj = lkk[j];
            sa0 -= s0[j]*kj;                       // a_t = -kk
            sa1 -= s1[j]*kj;
        }
        float o0 = 0.f, o1 = 0.f;
        #pragma unroll
        for (int j=0; j<NN; ++j){
            const float bj = lkk[j]*la[j];         // b_t = kk*a
            const float wj = lw[j], kj = lk[j], rj = lr[j];
            const float t0 = s0[j]*wj + sa0*bj + vi0*kj;
            const float t1 = s1[j]*wj + sa1*bj + vi1*kj;
            s0[j] = t0; o0 += t0*rj;
            s1[j] = t1; o1 += t1*rj;
        }
        y[g+i]    = o0;
        y[g+i+32] = o1;
        __syncthreads();
    }
}

// ---------------------------------------------------------------- GN + residual + gate
__global__ void gnorm_gate(const float* __restrict__ y, const float* __restrict__ r,
                           const float* __restrict__ k, const float* __restrict__ v,
                           const float* __restrict__ gbuf, const float* __restrict__ r_k,
                           const float* __restrict__ ln_g, const float* __restrict__ ln_b,
                           bf16* __restrict__ yg)
{
    __shared__ float red[NN], red2[NN];
    const int mh = blockIdx.x;
    const int m  = mh >> 4, h = mh & 15;
    const int tid = threadIdx.x;
    const int c   = h*NN + tid;
    const size_t g = (size_t)m*CC + c;
    const float yv = y[g];
    red[tid]  = yv;
    red2[tid] = r[g]*k[g]*r_k[c];
    __syncthreads();
    for (int s=32; s>0; s>>=1){
        if (tid<s){ red[tid]+=red[tid+s]; red2[tid]+=red2[tid+s]; }
        __syncthreads();
    }
    const float mu_ = red[0]*(1.f/NN);
    const float rk  = red2[0];
    __syncthreads();
    const float d0 = yv - mu_;
    red[tid] = d0*d0;
    __syncthreads();
    for (int s=32; s>0; s>>=1){ if (tid<s) red[tid]+=red[tid+s]; __syncthreads(); }
    const float var = red[0]*(1.f/NN);
    float yn = d0*rsqrtf(var + EPSV)*ln_g[c] + ln_b[c];
    yn += rk * v[g];
    yg[g] = (bf16)(yn * gbuf[g]);
}

// ---------------------------------------------------------------- launch
extern "C" void kernel_launch(void* const* d_in, const int* in_sizes, int n_in,
                              void* d_out, int out_size, void* d_ws, size_t ws_size,
                              hipStream_t stream) {
    const float* x      = (const float*)d_in[0];
    const float* vfirst = (const float*)d_in[1];
    const float* state0 = (const float*)d_in[2];
    const float* x_r = (const float*)d_in[3];
    const float* x_w = (const float*)d_in[4];
    const float* x_k = (const float*)d_in[5];
    const float* x_v = (const float*)d_in[6];
    const float* x_a = (const float*)d_in[7];
    const float* x_g = (const float*)d_in[8];
    const float* w0  = (const float*)d_in[9];
    const float* w1  = (const float*)d_in[10];
    const float* w2  = (const float*)d_in[11];
    const float* a0  = (const float*)d_in[12];
    const float* a1  = (const float*)d_in[13];
    const float* a2  = (const float*)d_in[14];
    const float* v0  = (const float*)d_in[15];
    const float* v1  = (const float*)d_in[16];
    const float* v2  = (const float*)d_in[17];
    const float* g1  = (const float*)d_in[18];
    const float* g2  = (const float*)d_in[19];
    const float* k_k = (const float*)d_in[20];
    const float* k_a = (const float*)d_in[21];
    const float* r_k = (const float*)d_in[22];
    const float* Wr  = (const float*)d_in[23];
    const float* Wk  = (const float*)d_in[24];
    const float* Wv  = (const float*)d_in[25];
    const float* Wo  = (const float*)d_in[26];
    const float* ln_g= (const float*)d_in[27];
    const float* ln_b= (const float*)d_in[28];

    char* ws = (char*)d_ws;
    size_t off = 0;
    auto alloc = [&](size_t bytes)->void*{
        void* p = ws + off; off = (off + bytes + 255) & ~(size_t)255; return p;
    };
    const size_t MC = (size_t)MM*CC;
    bf16* Wr_b = (bf16*)alloc(CC*(size_t)CC*2);
    bf16* Wk_b = (bf16*)alloc(CC*(size_t)CC*2);
    bf16* Wv_b = (bf16*)alloc(CC*(size_t)CC*2);
    bf16* Wo_b = (bf16*)alloc(CC*(size_t)CC*2);
    float* rb   = (float*)alloc(MC*4);
    float* kb_  = (float*)alloc(MC*4);
    float* vb   = (float*)alloc(MC*4);
    float* wdb  = (float*)alloc(MC*4);
    float* ab   = (float*)alloc(MC*4);
    float* gb   = (float*)alloc(MC*4);
    float* kkb  = (float*)alloc(MC*4);
    float* yb   = (float*)alloc(MC*4);
    float* vgate= (float*)alloc(MC*4);
    float* hid  = (float*)alloc((size_t)MM*128*4);
    bf16*  ygb  = (bf16*)alloc(MC*2);

    const int nW = CC*CC;
    cvt_f32_bf16<<<(nW+255)/256,256,0,stream>>>(Wr, Wr_b, nW);
    cvt_f32_bf16<<<(nW+255)/256,256,0,stream>>>(Wk, Wk_b, nW);
    cvt_f32_bf16<<<(nW+255)/256,256,0,stream>>>(Wv, Wv_b, nW);
    cvt_f32_bf16<<<(nW+255)/256,256,0,stream>>>(Wo, Wo_b, nW);

    dim3 gg(CC/64, MM/256);   // (16, 16), 256 threads/block
    wmma_gemm<0><<<gg,256,0,stream>>>(x, x_r, nullptr, Wr_b, rb);
    wmma_gemm<0><<<gg,256,0,stream>>>(x, x_k, nullptr, Wk_b, kb_);
    wmma_gemm<0><<<gg,256,0,stream>>>(x, x_v, nullptr, Wv_b, vb);

    dim3 g2d(CC/256, MM);
    lora_stage1<1><<<MM, 64,0,stream>>>(x, x_w, w1, 64,  hid);
    lora_stage2<2><<<g2d,256,0,stream>>>(hid, 64,  w2, w0, wdb);
    lora_stage1<0><<<MM, 64,0,stream>>>(x, x_a, a1, 64,  hid);
    lora_stage2<1><<<g2d,256,0,stream>>>(hid, 64,  a2, a0, ab);
    lora_stage1<0><<<MM, 32,0,stream>>>(x, x_v, v1, 32,  hid);
    lora_stage2<1><<<g2d,256,0,stream>>>(hid, 32,  v2, v0, vgate);
    lora_stage1<2><<<MM,128,0,stream>>>(x, x_g, g1, 128, hid);
    lora_stage2<0><<<g2d,256,0,stream>>>(hid, 128, g2, nullptr, gb);

    kv_prep<<<MM*HH,64,0,stream>>>(kb_, vb, ab, vgate, vfirst, k_k, k_a, kkb);
    rwkv_scan<<<BB*HH,32,0,stream>>>(rb, wdb, kb_, kkb, ab, vb, state0, yb);
    gnorm_gate<<<MM*HH,64,0,stream>>>(yb, rb, kb_, vb, gb, r_k, ln_g, ln_b, ygb);

    wmma_gemm<1><<<gg,256,0,stream>>>(nullptr, nullptr, ygb, Wo_b, (float*)d_out);

    hipMemcpyAsync((float*)d_out + MC, vfirst, MC*sizeof(float),
                   hipMemcpyDeviceToDevice, stream);
}
